// POCML_64742337020618
// MI455X (gfx1250) — matrix operand order repaired and checked by reading, not compiled, not run on hardware
//
#include <hip/hip_runtime.h>
#include <hip/hip_bf16.h>
#include <math.h>

// ---------------------------------------------------------------------------
// Problem constants (from reference)
// ---------------------------------------------------------------------------
#define N_OBS      20000
#define N_STATES   1024
#define N_ACTIONS  64
#define STATE_DIM  256
#define RFD        2048
#define DECAY      0.9f
#define BETA_OBS   1.0f
#define BETA_STATE 1.0f

typedef __bf16 bf16;
typedef __bf16 v8bf  __attribute__((ext_vector_type(8)));
typedef __bf16 v16bf __attribute__((ext_vector_type(16)));
typedef float  v8f   __attribute__((ext_vector_type(8)));

// ---------------------------------------------------------------------------
// Block reduction helpers
// ---------------------------------------------------------------------------
__device__ __forceinline__ float block_sum(float v, float* sdata) {
    int tid = threadIdx.x;
    sdata[tid] = v;
    __syncthreads();
    for (int s = blockDim.x >> 1; s > 0; s >>= 1) {
        if (tid < s) sdata[tid] += sdata[tid + s];
        __syncthreads();
    }
    float r = sdata[0];
    __syncthreads();
    return r;
}

__device__ __forceinline__ float block_max(float v, float* sdata) {
    int tid = threadIdx.x;
    sdata[tid] = v;
    __syncthreads();
    for (int s = blockDim.x >> 1; s > 0; s >>= 1) {
        if (tid < s) sdata[tid] = fmaxf(sdata[tid], sdata[tid + s]);
        __syncthreads();
    }
    float r = sdata[0];
    __syncthreads();
    return r;
}

// ---------------------------------------------------------------------------
// K1: v = V @ action   (V: [256,64] row-major)
// ---------------------------------------------------------------------------
__global__ void k_v_gemv(const float* __restrict__ V, const float* __restrict__ action,
                         float* __restrict__ v) {
    int i = threadIdx.x;   // 256 threads, one per state-dim
    float acc = 0.f;
#pragma unroll
    for (int a = 0; a < N_ACTIONS; ++a)
        acc = fmaf(V[i * N_ACTIONS + a], action[a], acc);
    v[i] = acc;
}

// ---------------------------------------------------------------------------
// K2: theta[d] = W[d,:] @ v    (one block per d, coalesced row read)
// ---------------------------------------------------------------------------
__global__ void k_theta(const float* __restrict__ W, const float* __restrict__ v,
                        float* __restrict__ theta) {
    __shared__ float sdata[256];
    int d = blockIdx.x;
    float p = W[d * STATE_DIM + threadIdx.x] * v[threadIdx.x];
    float s = block_sum(p, sdata);
    if (threadIdx.x == 0) theta[d] = s;
}

// ---------------------------------------------------------------------------
// K3: state0 = M @ obs  (blocks: 2*RFD; even = re, odd = im). Streams 327 MB.
// ---------------------------------------------------------------------------
__global__ void k_state0(const float* __restrict__ M_re, const float* __restrict__ M_im,
                         const float* __restrict__ obs,
                         float* __restrict__ s0_re, float* __restrict__ s0_im) {
    __shared__ float sdata[256];
    int b = blockIdx.x;
    int d = b >> 1;
    const float* __restrict__ Mp = (b & 1) ? M_im : M_re;
    const float* __restrict__ row = Mp + (size_t)d * N_OBS;
    float acc = 0.f;
    for (int n = threadIdx.x; n < N_OBS; n += 256)
        acc = fmaf(row[n], obs[n], acc);
    float s = block_sum(acc, sdata);
    if (threadIdx.x == 0) {
        if (b & 1) s0_im[d] = s; else s0_re[d] = s;
    }
}

// ---------------------------------------------------------------------------
// K4: state1 = state0 * exp(i*theta)
// ---------------------------------------------------------------------------
__global__ void k_state1(const float* __restrict__ theta,
                         const float* __restrict__ s0_re, const float* __restrict__ s0_im,
                         float* __restrict__ s1_re, float* __restrict__ s1_im) {
    int d = blockIdx.x * blockDim.x + threadIdx.x;
    float sn, cs;
    __sincosf(theta[d], &sn, &cs);
    float re = s0_re[d], im = s0_im[d];
    s1_re[d] = re * cs - im * sn;
    s1_im[d] = re * sn + im * cs;
}

// ---------------------------------------------------------------------------
// K5a/K5b: bf16 conversion.  W -> Wb (same layout [2048,256]);
//          Q [256,1024] -> Qt bf16 transposed [1024,256] (K contiguous for B frags)
// ---------------------------------------------------------------------------
__global__ void k_convW(const float* __restrict__ W, bf16* __restrict__ Wb) {
    int i = blockIdx.x * blockDim.x + threadIdx.x;
    Wb[i] = (bf16)W[i];
}
__global__ void k_convQ(const float* __restrict__ Q, bf16* __restrict__ Qt) {
    int gid = blockIdx.x * blockDim.x + threadIdx.x;
    int n = gid >> 8;      // state index 0..1023
    int k = gid & 255;     // K index     0..255
    Qt[n * STATE_DIM + k] = (bf16)Q[k * N_STATES + n];
}

// ---------------------------------------------------------------------------
// K6: WMMA GEMM  A = W @ Q  (M=2048, N=1024, K=256) in bf16 -> f32 accum.
// One 16x16 output tile per wave, 8 waves per block, K-loop of 8
// v_wmma_f32_16x16x32_bf16.  Result (angle matrix) stored TRANSPOSED:
// A_T[s*2048 + d]  so both downstream GEMVs read coalesced.
//
// A-frag layout (ISA 7.12.2, 16-bit A 16x32): lane row m = lane&15,
//   halves 0..7  -> K = k0..k0+7,  halves 8..15 -> K = k0+16..k0+23,
//   k0 = ktile*32 + 8*(lane>=16).
// B-frag layout (32x16): lane col n = lane&15,
//   halves 0..7 -> K = kb..kb+7,  halves 8..15 -> K = kb+16..kb+23, kb = ktile*32 + 8*(lane>=16).
// C/D layout: lane col n = lane&15; VGPR r -> row m = r + 8*(lane>=16).
// ---------------------------------------------------------------------------
__global__ void k_wmma_angles(const bf16* __restrict__ Wb, const bf16* __restrict__ Qt,
                              float* __restrict__ A_T) {
    const int lane = threadIdx.x & 31;
    const int wave = threadIdx.x >> 5;
    const int tile = blockIdx.x * 8 + wave;       // 8192 tiles total
    const int mtile = tile >> 6;                  // 128 M-tiles
    const int ntile = tile & 63;                  // 64  N-tiles

    const int hi   = (lane & 16) ? 1 : 0;
    const int m    = mtile * 16 + (lane & 15);
    const int n    = ntile * 16 + (lane & 15);

    v8f c = {};
#pragma unroll
    for (int kt = 0; kt < 8; ++kt) {
        const int k0 = kt * 32 + hi * 8;          // A frag K base
        const int kb = kt * 32 + hi * 8;          // B frag K base
        v8bf alo = *reinterpret_cast<const v8bf*>(Wb + m * STATE_DIM + k0);
        v8bf ahi = *reinterpret_cast<const v8bf*>(Wb + m * STATE_DIM + k0 + 16);
        v8bf blo = *reinterpret_cast<const v8bf*>(Qt + n * STATE_DIM + kb);
        v8bf bhi = *reinterpret_cast<const v8bf*>(Qt + n * STATE_DIM + kb + 16);
        v16bf a = __builtin_shufflevector(alo, ahi, 0,1,2,3,4,5,6,7,8,9,10,11,12,13,14,15);
        v16bf b = __builtin_shufflevector(blo, bhi, 0,1,2,3,4,5,6,7,8,9,10,11,12,13,14,15);
        c = __builtin_amdgcn_wmma_f32_16x16x32_bf16(
                /*neg_a=*/false, a, /*neg_b=*/false, b,
                /*c_mod=*/(short)0, c, /*reuse_a=*/false, /*reuse_b=*/false);
    }

    // store transposed: per lane, its 8 C values are m-contiguous -> 2x b128
    const int mbase = mtile * 16 + hi * 8;
    float* dst = A_T + (size_t)n * RFD + mbase;
    *reinterpret_cast<float4*>(dst)     = make_float4(c[0], c[1], c[2], c[3]);
    *reinterpret_cast<float4*>(dst + 4) = make_float4(c[4], c[5], c[6], c[7]);
}

// ---------------------------------------------------------------------------
// K7: score[s] = (1/D) * sum_d cos(A[d,s])*s1_re[d] + sin(A[d,s])*s1_im[d]
// one block per state s; A_T row read is coalesced.
// ---------------------------------------------------------------------------
__global__ void k_score(const float* __restrict__ A_T,
                        const float* __restrict__ s1_re, const float* __restrict__ s1_im,
                        float* __restrict__ score) {
    __shared__ float sdata[256];
    int s = blockIdx.x;
    const float* __restrict__ row = A_T + (size_t)s * RFD;
    float acc = 0.f;
    for (int d = threadIdx.x; d < RFD; d += 256) {
        float sn, cs;
        __sincosf(row[d], &sn, &cs);
        acc = fmaf(cs, s1_re[d], acc);
        acc = fmaf(sn, s1_im[d], acc);
    }
    float t = block_sum(acc, sdata);
    if (threadIdx.x == 0) score[s] = (BETA_STATE / (float)RFD) * t;
}

// ---------------------------------------------------------------------------
// K8: weights = softmax(score)   (single block, 1024 threads)
// ---------------------------------------------------------------------------
__global__ void k_weights(const float* __restrict__ score, float* __restrict__ weights) {
    __shared__ float sdata[1024];
    int s = threadIdx.x;
    float x = score[s];
    float mx = block_max(x, sdata);
    float e = __expf(x - mx);
    float sum = block_sum(e, sdata);
    weights[s] = e / sum;
}

// ---------------------------------------------------------------------------
// K9: state2[d] = sum_s exp(i*A[d,s]) * w[s]  ;  weights staged in LDS
// ---------------------------------------------------------------------------
__global__ void k_state2(const float* __restrict__ A_T, const float* __restrict__ weights,
                         float* __restrict__ s2_re, float* __restrict__ s2_im) {
    __shared__ float wsm[N_STATES];
    for (int i = threadIdx.x; i < N_STATES; i += 256) wsm[i] = weights[i];
    __syncthreads();
    int d = blockIdx.x * blockDim.x + threadIdx.x;
    float are = 0.f, aim = 0.f;
    for (int s = 0; s < N_STATES; ++s) {
        float sn, cs;
        __sincosf(A_T[(size_t)s * RFD + d], &sn, &cs);  // coalesced in d
        float w = wsm[s];
        are = fmaf(cs, w, are);
        aim = fmaf(sn, w, aim);
    }
    s2_re[d] = are;
    s2_im[d] = aim;
}

// ---------------------------------------------------------------------------
// K10: dot = Re(state0 . conj(state2))   (single block)
// ---------------------------------------------------------------------------
__global__ void k_dot(const float* __restrict__ s0_re, const float* __restrict__ s0_im,
                      const float* __restrict__ s2_re, const float* __restrict__ s2_im,
                      float* __restrict__ dotbuf) {
    __shared__ float sdata[256];
    float acc = 0.f;
    for (int d = threadIdx.x; d < RFD; d += 256)
        acc += s0_re[d] * s2_re[d] + s0_im[d] * s2_im[d];
    float t = block_sum(acc, sdata);
    if (threadIdx.x == 0) dotbuf[0] = t;
}

// ---------------------------------------------------------------------------
// K11: obs_score[n] = (1/D)*(DECAY*(M_re[:,n].s2_re + M_im[:,n].s2_im) + obs[n]*dot)
// Second (and last) streaming pass over M; s2 vectors staged in LDS.
// ---------------------------------------------------------------------------
__global__ void k_obs_score(const float* __restrict__ M_re, const float* __restrict__ M_im,
                            const float* __restrict__ obs,
                            const float* __restrict__ s2_re, const float* __restrict__ s2_im,
                            const float* __restrict__ dotbuf,
                            float* __restrict__ obs_score) {
    __shared__ float lre[RFD];
    __shared__ float lim[RFD];
    for (int i = threadIdx.x; i < RFD; i += 256) { lre[i] = s2_re[i]; lim[i] = s2_im[i]; }
    __syncthreads();

    int n = blockIdx.x * blockDim.x + threadIdx.x;
    if (n >= N_OBS) return;
    float acc = 0.f;
#pragma unroll 4
    for (int d = 0; d < RFD; ++d) {
        acc = fmaf(M_re[(size_t)d * N_OBS + n], lre[d], acc);  // coalesced across lanes
        acc = fmaf(M_im[(size_t)d * N_OBS + n], lim[d], acc);
    }
    obs_score[n] = (BETA_OBS / (float)RFD) * (DECAY * acc + obs[n] * dotbuf[0]);
}

// ---------------------------------------------------------------------------
// K12: out = softmax(obs_score) over 20000   (single block, 1024 threads)
// ---------------------------------------------------------------------------
__global__ void k_softmax_out(const float* __restrict__ obs_score, float* __restrict__ out) {
    __shared__ float sdata[1024];
    float mx = -INFINITY;
    for (int n = threadIdx.x; n < N_OBS; n += 1024) mx = fmaxf(mx, obs_score[n]);
    mx = block_max(mx, sdata);
    float sum = 0.f;
    for (int n = threadIdx.x; n < N_OBS; n += 1024) sum += __expf(obs_score[n] - mx);
    sum = block_sum(sum, sdata);
    float inv = 1.0f / sum;
    for (int n = threadIdx.x; n < N_OBS; n += 1024)
        out[n] = __expf(obs_score[n] - mx) * inv;
}

// ---------------------------------------------------------------------------
// Launcher
// ---------------------------------------------------------------------------
extern "C" void kernel_launch(void* const* d_in, const int* in_sizes, int n_in,
                              void* d_out, int out_size, void* d_ws, size_t ws_size,
                              hipStream_t stream) {
    const float* Q      = (const float*)d_in[0];  // [256,1024]
    const float* V      = (const float*)d_in[1];  // [256,64]
    const float* W      = (const float*)d_in[2];  // [2048,256]
    const float* M_re   = (const float*)d_in[3];  // [2048,20000]
    const float* M_im   = (const float*)d_in[4];  // [2048,20000]
    const float* obs    = (const float*)d_in[5];  // [20000]
    const float* action = (const float*)d_in[6];  // [64]
    float* out = (float*)d_out;                   // [20000]

    // workspace layout (floats, then bf16, then f32 angle matrix)
    float* wf = (float*)d_ws;
    float* v         = wf;                 // 256
    float* theta     = wf + 256;           // 2048
    float* s0_re     = wf + 2304;          // 2048
    float* s0_im     = wf + 4352;          // 2048
    float* s1_re     = wf + 6400;          // 2048
    float* s1_im     = wf + 8448;          // 2048
    float* s2_re     = wf + 10496;         // 2048
    float* s2_im     = wf + 12544;         // 2048
    float* score     = wf + 14592;         // 1024
    float* weights   = wf + 15616;         // 1024
    float* dotbuf    = wf + 16640;         // 16 (use [0])
    float* obs_score = wf + 16656;         // 20000  -> ends at 36656
    bf16*  Wb  = (bf16*)(wf + 36672);                       // 2048*256 bf16 (1 MB)
    bf16*  Qt  = (bf16*)((char*)Wb + (size_t)RFD * STATE_DIM * sizeof(bf16)); // 512 KB
    float* A_T = (float*)((char*)Qt + (size_t)N_STATES * STATE_DIM * sizeof(bf16)); // 8 MB

    k_v_gemv<<<1, 256, 0, stream>>>(V, action, v);
    k_theta<<<RFD, 256, 0, stream>>>(W, v, theta);
    k_state0<<<2 * RFD, 256, 0, stream>>>(M_re, M_im, obs, s0_re, s0_im);
    k_state1<<<RFD / 256, 256, 0, stream>>>(theta, s0_re, s0_im, s1_re, s1_im);
    k_convW<<<(RFD * STATE_DIM) / 256, 256, 0, stream>>>(W, Wb);
    k_convQ<<<(N_STATES * STATE_DIM) / 256, 256, 0, stream>>>(Q, Qt);
    k_wmma_angles<<<(128 * 64) / 8, 256, 0, stream>>>(Wb, Qt, A_T);
    k_score<<<N_STATES, 256, 0, stream>>>(A_T, s1_re, s1_im, score);
    k_weights<<<1, 1024, 0, stream>>>(score, weights);
    k_state2<<<RFD / 256, 256, 0, stream>>>(A_T, weights, s2_re, s2_im);
    k_dot<<<1, 256, 0, stream>>>(s0_re, s0_im, s2_re, s2_im, dotbuf);
    k_obs_score<<<(N_OBS + 255) / 256, 256, 0, stream>>>(M_re, M_im, obs, s2_re, s2_im,
                                                         dotbuf, obs_score);
    k_softmax_out<<<1, 1024, 0, stream>>>(obs_score, out);
    (void)in_sizes; (void)n_in; (void)out_size; (void)ws_size;
}
